// AttentionS2_21337397526732
// MI455X (gfx1250) — matrix operand order repaired and checked by reading, not compiled, not run on hardware
//
#include <hip/hip_runtime.h>

#define CIN   512
#define HW    4096
#define HEADS 8
#define HD    64
#define LOG2E 1.44269504088896340736f

typedef __attribute__((ext_vector_type(16))) _Float16 v16h;
typedef __attribute__((ext_vector_type(8)))  float    v8f;

// ---------------------------------------------------------------------------
// Fragment-friendly ("FF") tile layouts, 512 halfs per 16x32 / 32x16 tile,
// stored as tile[lane][e] so a lane loads its 16 WMMA elements as one 32B read.
//
// A-tile (16 rows x K=32), contiguous-k mapping:
//   lane = ahi*16 + row,  K = 16*ahi + e
//   inverse: ahi = k>>4 ; e = k&15
// B-tile (K=32 rows x 16 cols):
//   lane = (row>=16)*16 + col, e = row&15
// ---------------------------------------------------------------------------

// ---------------------------------------------------------------------------
// Kernel 0a: stage X (fp32 [c][t]) into fp16 FF B-tiles. One wave per tile.
// ---------------------------------------------------------------------------
__global__ void __launch_bounds__(256)
swizzle_x_kernel(const float* __restrict__ x, _Float16* __restrict__ Xs)
{
    const int wid  = blockIdx.x * 8 + (threadIdx.x >> 5);
    const int lane = threadIdx.x & 31;
    const int half = lane >> 4, ln = lane & 15;
    const int ks = wid >> 8;                    // k-slice (32 channels)
    const int tt = wid & 255;                   // 16-token tile
    const int kk = ks * 32, t = tt * 16 + ln;

    v16h frag;
    #pragma unroll
    for (int e = 0; e < 16; ++e)
        frag[e] = (_Float16)x[(kk + e + half * 16) * HW + t];
    *(v16h*)(Xs + (size_t)(ks * 256 + tt) * 512 + lane * 16) = frag;
}

// ---------------------------------------------------------------------------
// Kernel 0b: stage the 4 weight matrices (fp32 512x512) into fp16 FF A-tiles,
// so GEMM A-operands are single 32B loads with no in-loop conversion.
// Tile grid per matrix: 32 o-tiles x 16 k-slices. One wave per tile.
// ---------------------------------------------------------------------------
__global__ void __launch_bounds__(256)
swizzle_w_kernel(const float* __restrict__ qw, const float* __restrict__ kw,
                 const float* __restrict__ vw, const float* __restrict__ pw,
                 _Float16* __restrict__ Wff)
{
    const int wid  = blockIdx.x * 8 + (threadIdx.x >> 5);
    const int lane = threadIdx.x & 31;
    const int hi   = lane >> 4, ln = lane & 15;
    const int mat  = wid >> 9;                  // 4 matrices
    const int rem  = wid & 511;
    const int ot   = rem >> 4;                  // o-tile
    const int ks   = rem & 15;                  // k-slice
    const float* w = (mat == 0) ? qw : (mat == 1) ? kw : (mat == 2) ? vw : pw;

    // A layout: one contiguous 16-float run per lane at kk+16*hi
    const float* wr = &w[(ot * 16 + ln) * CIN + ks * 32 + 16 * hi];
    float tf[16];
    *(float4*)(tf + 0)  = *(const float4*)(wr + 0);
    *(float4*)(tf + 4)  = *(const float4*)(wr + 4);
    *(float4*)(tf + 8)  = *(const float4*)(wr + 8);
    *(float4*)(tf + 12) = *(const float4*)(wr + 12);
    v16h a;
    #pragma unroll
    for (int e = 0; e < 16; ++e) a[e] = (_Float16)tf[e];
    *(v16h*)(Wff + (size_t)((mat * 32 + ot) * 16 + ks) * 512 + lane * 16) = a;
}

// ---------------------------------------------------------------------------
// Kernel 1: fused Q/K/V projections.  Y(512x4096) = W(512x512) * X(512x4096).
// Wave computes 16(o) x 64(t); A and B are single v16h FF loads.
// Results scattered (2B stores) into FF layouts:
//   Qff: A-tiles  (head, qtile, dslice)        for S = Q K^T
//   Kff: B-tiles  (head, dslice, ktile)        for S = Q K^T
//   Vff: B-tiles  (head, kb32, dtile)          for O = P V
// ---------------------------------------------------------------------------
__global__ void __launch_bounds__(256)
qkv_proj_kernel(const _Float16* __restrict__ Xs, const _Float16* __restrict__ Wff,
                const float* __restrict__ qb, const float* __restrict__ kb,
                const float* __restrict__ vb,
                _Float16* __restrict__ Qff, _Float16* __restrict__ Kff,
                _Float16* __restrict__ Vff)
{
    const int wid  = blockIdx.x * 8 + (threadIdx.x >> 5);
    const int lane = threadIdx.x & 31;
    const int hi   = lane >> 4, ln = lane & 15;

    const int mat = wid / 2048;                 // 0=Q 1=K 2=V
    const int rem = wid % 2048;
    const int o0  = (rem >> 6) << 4;
    const int t0  = (rem & 63) << 6;

    const float* bias = (mat == 0) ? qb : (mat == 1) ? kb : vb;
    const _Float16* wff = Wff + ((size_t)(mat * 32 + (o0 >> 4)) * 16) * 512;

    v8f acc[4] = {};
    for (int ks = 0; ks < 16; ++ks) {
        const v16h a = *(const v16h*)(wff + (size_t)ks * 512 + lane * 16);
        #pragma unroll
        for (int j = 0; j < 4; ++j) {
            const v16h b = *(const v16h*)(Xs + (size_t)(ks * 256 + (t0 >> 4) + j) * 512 + lane * 16);
            acc[j] = __builtin_amdgcn_wmma_f32_16x16x32_f16(
                         false, a, false, b, (short)0, acc[j], false, false);
        }
    }

    #pragma unroll
    for (int r = 0; r < 8; ++r) {
        const int o    = o0 + r + hi * 8;
        const int head = o >> 6, d = o & 63;
        const float bv = bias[o];
        #pragma unroll
        for (int j = 0; j < 4; ++j) {
            const int t = t0 + j * 16 + ln;
            const _Float16 val = (_Float16)(acc[j][r] + bv);
            if (mat == 0) {                      // Q -> A-tile layout
                const int ds = d >> 5, d31 = d & 31;
                const int ahi = d31 >> 4;
                const int ae  = d31 & 15;
                const int lc  = ahi * 16 + (t & 15);
                Qff[(size_t)(((head * 256 + (t >> 4)) * 2) + ds) * 512 + lc * 16 + ae] = val;
            } else if (mat == 1) {               // K -> B-tile layout (rows=d)
                const int ds = d >> 5, d31 = d & 31;
                const int lc = (d31 >> 4) * 16 + (t & 15);
                Kff[(size_t)(((head * 2 + ds) * 256) + (t >> 4)) * 512 + lc * 16 + (d31 & 15)] = val;
            } else {                             // V -> B-tile layout (rows=t)
                const int lc = (((t & 31) >> 4)) * 16 + (d & 15);
                Vff[(size_t)(((head * 128 + (t >> 5)) * 4) + (d >> 4)) * 512 + lc * 16 + (t & 15)] = val;
            }
        }
    }
}

// ---------------------------------------------------------------------------
// Kernel 2: flash attention, one wave per (head, 16-query strip).
// Per 32-key block: 4 WMMA for S, online softmax (base-2, log2e folded into
// scale/bias) with 16-lane shuffle reductions, P scattered into LDS in A-tile
// lane-major order (ds_store_b16), reloaded as one 32B ds read, 4 WMMA for
// O += P V.  Output scattered fp16 into FF B-tiles for the out-projection.
// ---------------------------------------------------------------------------
__global__ void __launch_bounds__(256)
attn_kernel(const _Float16* __restrict__ Qff, const _Float16* __restrict__ Kff,
            const _Float16* __restrict__ Vff, const float* __restrict__ lqw,
            _Float16* __restrict__ Off)
{
    __shared__ __align__(32) _Float16 lds[8 * 512];   // one A-tile per wave

    const int warp = threadIdx.x >> 5;
    const int wid  = blockIdx.x * 8 + warp;
    const int lane = threadIdx.x & 31;
    const int hi   = lane >> 4, ln = lane & 15;

    const int head = wid >> 8;
    const int qt   = wid & 255;
    const int q0   = qt << 4;
    const float scale = 0.125f * LOG2E;               // (1/sqrt(64)) * log2(e)

    // resident Q fragments (single 32B loads from pre-swizzled A-tiles)
    const v16h a0 = *(const v16h*)(Qff + (size_t)((head * 256 + qt) * 2 + 0) * 512 + lane * 16);
    const v16h a1 = *(const v16h*)(Qff + (size_t)((head * 256 + qt) * 2 + 1) * 512 + lane * 16);

    float mrow[8], lrow[8];
    v8f   oacc[4] = {};
    #pragma unroll
    for (int r = 0; r < 8; ++r) { mrow[r] = -3.0e38f; lrow[r] = 0.0f; }

    _Float16* myLds = lds + warp * 512;
    const _Float16* Kh0 = Kff + (size_t)(head * 2 + 0) * 256 * 512;
    const _Float16* Kh1 = Kff + (size_t)(head * 2 + 1) * 256 * 512;

    for (int kb = 0; kb < HW; kb += 32) {
        const int kt = kb >> 4;
        // ---- S = Q K^T (16q x 32k) ----
        v8f s0 = {}, s1 = {};
        {
            v16h b;
            b = *(const v16h*)(Kh0 + (size_t)kt * 512 + lane * 16);
            s0 = __builtin_amdgcn_wmma_f32_16x16x32_f16(false, a0, false, b, (short)0, s0, false, false);
            b = *(const v16h*)(Kh1 + (size_t)kt * 512 + lane * 16);
            s0 = __builtin_amdgcn_wmma_f32_16x16x32_f16(false, a1, false, b, (short)0, s0, false, false);
            b = *(const v16h*)(Kh0 + (size_t)(kt + 1) * 512 + lane * 16);
            s1 = __builtin_amdgcn_wmma_f32_16x16x32_f16(false, a0, false, b, (short)0, s1, false, false);
            b = *(const v16h*)(Kh1 + (size_t)(kt + 1) * 512 + lane * 16);
            s1 = __builtin_amdgcn_wmma_f32_16x16x32_f16(false, a1, false, b, (short)0, s1, false, false);
        }

        // additive log-quadrature bias (converted to base-2 domain)
        const float bias0 = lqw[kb + ln] * LOG2E;
        const float bias1 = lqw[kb + 16 + ln] * LOG2E;

        if (kb + 32 < HW) {                    // pull next K tiles toward caches
            __builtin_prefetch(Kh0 + (size_t)(kt + 2) * 512 + lane * 16, 0, 1);
            __builtin_prefetch(Kh1 + (size_t)(kt + 2) * 512 + lane * 16, 0, 1);
        }

        // ---- online softmax in base-2; scatter P into consumer A-tile order ----
        const int ahiL = (ln >> 3) & 1;        // unused; k=ln goes to lanes 0-15, k=16+ln to lanes 16-31
        #pragma unroll
        for (int r = 0; r < 8; ++r) {
            float v0 = s0[r] * scale + bias0;
            float v1 = s1[r] * scale + bias1;
            float rm = fmaxf(v0, v1);
            #pragma unroll
            for (int off = 8; off > 0; off >>= 1)
                rm = fmaxf(rm, __shfl_xor(rm, off, 32));
            const float mnew = fmaxf(mrow[r], rm);
            const float corr = exp2f(mrow[r] - mnew);
            mrow[r] = mnew;
            const float p0 = exp2f(v0 - mnew);
            const float p1 = exp2f(v1 - mnew);
            float rs = p0 + p1;
            #pragma unroll
            for (int off = 8; off > 0; off >>= 1)
                rs += __shfl_xor(rs, off, 32);
            lrow[r] = lrow[r] * corr + rs;
            #pragma unroll
            for (int n = 0; n < 4; ++n) oacc[n][r] *= corr;

            const int lc = r + hi * 8;                      // consumer lane
            myLds[lc * 16 + ln]        = (_Float16)p0;      // k = ln
            myLds[(16 + lc) * 16 + ln] = (_Float16)p1;      // k = 16+ln
        }

        // reload P as A fragment: one 32B LDS read (same wave; DS ops in-order)
        const v16h pa = *(const v16h*)(myLds + lane * 16);

        // ---- O += P V ----
        const _Float16* vb = Vff + (size_t)(head * 128 + (kb >> 5)) * 4 * 512;
        #pragma unroll
        for (int n = 0; n < 4; ++n) {
            const v16h bv = *(const v16h*)(vb + (size_t)n * 512 + lane * 16);
            oacc[n] = __builtin_amdgcn_wmma_f32_16x16x32_f16(
                          false, pa, false, bv, (short)0, oacc[n], false, false);
        }
    }

    // normalize and scatter into FF B-tiles [i-slice][token-tile]
    #pragma unroll
    for (int n = 0; n < 4; ++n) {
        const int d = n * 16 + ln;
        const int i = head * HD + d;
        const int isl = i >> 5, ihalf = (i >> 4) & 1, ie = i & 15;
        #pragma unroll
        for (int r = 0; r < 8; ++r) {
            const int q  = q0 + r + hi * 8;
            const int lc = ihalf * 16 + (q & 15);
            Off[(size_t)(isl * 256 + (q >> 4)) * 512 + lc * 16 + ie] =
                (_Float16)(oacc[n][r] / lrow[r]);
        }
    }
}

// ---------------------------------------------------------------------------
// Kernel 3: output projection. out(512x4096 f32) = Pw(512x512) * O + b
// ---------------------------------------------------------------------------
__global__ void __launch_bounds__(256)
proj_out_kernel(const _Float16* __restrict__ Off, const _Float16* __restrict__ Wff,
                const float* __restrict__ pb, float* __restrict__ out)
{
    const int wid  = blockIdx.x * 8 + (threadIdx.x >> 5);
    const int lane = threadIdx.x & 31;
    const int hi   = lane >> 4, ln = lane & 15;
    const int o0   = (wid >> 6) << 4;
    const int t0   = (wid & 63) << 6;

    const _Float16* wff = Wff + ((size_t)(3 * 32 + (o0 >> 4)) * 16) * 512;

    v8f acc[4] = {};
    for (int isl = 0; isl < 16; ++isl) {
        const v16h a = *(const v16h*)(wff + (size_t)isl * 512 + lane * 16);
        #pragma unroll
        for (int j = 0; j < 4; ++j) {
            const v16h b = *(const v16h*)(Off + (size_t)(isl * 256 + (t0 >> 4) + j) * 512 + lane * 16);
            acc[j] = __builtin_amdgcn_wmma_f32_16x16x32_f16(
                         false, a, false, b, (short)0, acc[j], false, false);
        }
    }

    #pragma unroll
    for (int r = 0; r < 8; ++r) {
        const int oc = o0 + r + hi * 8;
        const float bv = pb[oc];
        #pragma unroll
        for (int j = 0; j < 4; ++j)
            out[oc * HW + t0 + j * 16 + ln] = acc[j][r] + bv;
    }
}

// ---------------------------------------------------------------------------
extern "C" void kernel_launch(void* const* d_in, const int* in_sizes, int n_in,
                              void* d_out, int out_size, void* d_ws, size_t ws_size,
                              hipStream_t stream) {
    const float* x   = (const float*)d_in[0];
    const float* qw  = (const float*)d_in[1];
    const float* qb  = (const float*)d_in[2];
    const float* kw  = (const float*)d_in[3];
    const float* kb  = (const float*)d_in[4];
    const float* vw  = (const float*)d_in[5];
    const float* vb  = (const float*)d_in[6];
    const float* pw  = (const float*)d_in[7];
    const float* pb  = (const float*)d_in[8];
    const float* lqw = (const float*)d_in[9];

    // fp16 workspace, all fragment-friendly layouts: 5x4MB tensors + 2MB weights
    const size_t TEN = (size_t)CIN * HW;        // 2M halfs per activation tensor
    _Float16* Xs  = (_Float16*)d_ws;
    _Float16* Qff = Xs  + TEN;
    _Float16* Kff = Qff + TEN;
    _Float16* Vff = Kff + TEN;
    _Float16* Off = Vff + TEN;
    _Float16* Wff = Off + TEN;                  // 4 * 32*16*512 halfs = 1M halfs

    swizzle_x_kernel<<<512, 256, 0, stream>>>(x, Xs);                          // 4096 waves
    swizzle_w_kernel<<<256, 256, 0, stream>>>(qw, kw, vw, pw, Wff);            // 2048 waves
    qkv_proj_kernel<<<768, 256, 0, stream>>>(Xs, Wff, qb, kb, vb,
                                             Qff, Kff, Vff);                   // 6144 waves
    attn_kernel<<<256, 256, 0, stream>>>(Qff, Kff, Vff, lqw, Off);             // 2048 waves
    proj_out_kernel<<<256, 256, 0, stream>>>(Off, Wff, pb, (float*)d_out);     // 2048 waves
}
